// Past_transformer_67628555043091
// MI455X (gfx1250) — matrix operand order repaired and checked by reading, not compiled, not run on hardware
//
#include <hip/hip_runtime.h>
#include <math.h>

typedef __attribute__((ext_vector_type(16))) _Float16 v16h;
typedef __attribute__((ext_vector_type(8)))  _Float16 v8h;
typedef __attribute__((ext_vector_type(8)))  float    v8f;

namespace {
constexpr int kD    = 512;    // d_model
constexpr int kH    = 8;      // heads
constexpr int kDK   = 64;     // head dim
constexpr int kDFF  = 2048;   // ffn dim
constexpr int kE    = 3;      // encoder layers
constexpr int kLenc = 1024;
constexpr int kLdec = 512;
constexpr int kB    = 8;
constexpr int kPred = 256;
constexpr int kCout = 64;
}

__device__ __forceinline__ float warp_sum(float v) {
#pragma unroll
  for (int off = 16; off > 0; off >>= 1) v += __shfl_xor(v, off, 32);
  return v;
}
__device__ __forceinline__ float warp_max(float v) {
#pragma unroll
  for (int off = 16; off > 0; off >>= 1) v = fmaxf(v, __shfl_xor(v, off, 32));
  return v;
}
__device__ __forceinline__ float gelu_exact(float x) {
  return 0.5f * x * (1.0f + erff(x * 0.70710678118654752440f));
}

// ---------------- conversion kernels ----------------
__global__ void k_cvt_f16(const float* __restrict__ src, _Float16* __restrict__ dst, int n) {
  int i = blockIdx.x * blockDim.x + threadIdx.x;
  if (i < n) dst[i] = (_Float16)src[i];
}

// W [K,N] f32 row-major  ->  Wt [N,K] f16 row-major (B^T layout for WMMA B loads)
__global__ void k_cvt_transpose(const float* __restrict__ W, _Float16* __restrict__ Wt,
                                int K, int N) {
  int i = blockIdx.x * blockDim.x + threadIdx.x;
  if (i < K * N) {
    int k = i / N, n = i - k * N;
    Wt[(size_t)n * K + k] = (_Float16)W[i];
  }
}

// ---------------- WMMA GEMM ----------------
// C[M,N] = A[M,K] * B[K,N] (+bias) with B supplied transposed (Bt[N,K] row-major).
// One wave computes a 32x64 tile: 2 M-subtiles x 4 N-subtiles, 8 f32 accumulators.
// Explicit ping-pong double buffering over K (K % 64 == 0 for every GEMM in this
// net): fragments for stage p^1 are loaded before the WMMAs consuming stage p,
// so they are live across the WMMAs -> distinct registers + staged waits.
// grid = (N/64, M/32, z-batch).
// ISA 7.12.2 layouts:
//   A frag  : lane l holds row (l&15); f16 elems [0..7]=K kb+8h.., [8..15]=K kb+16+8h.. (h=l>>4)
//   B frag  : lane l holds col n=(l&15); elems [0..7]=K kb+8h.., [8..15]=K kb+16+8h.. (same as A)
//   C/D     : lane l col n=(l&15); vgpr r row 8h+r
__global__ __launch_bounds__(32)
void k_wmma_gemm(const _Float16* __restrict__ A, long long aB, int lda,
                 const _Float16* __restrict__ Bt, long long bB, int ldbt,
                 const float* __restrict__ bias,
                 float* __restrict__ Cf, long long cB, int ldc,
                 _Float16* __restrict__ Ch, long long chB, int ldch, int transCh,
                 int K, int act) {
  const int lane = threadIdx.x & 31;
  const int nl = lane & 15;
  const int hb = lane >> 4;
  const int m0 = blockIdx.y * 32;
  const int n0 = blockIdx.x * 64;
  const long long z = blockIdx.z;

  const _Float16* Az = A + z * aB;
  const _Float16* Btz = Bt + z * bB;

  v8f acc[2][4];
#pragma unroll
  for (int mi = 0; mi < 2; ++mi)
#pragma unroll
    for (int t = 0; t < 4; ++t) acc[mi][t] = v8f{};

  const _Float16* arow0 = Az + (size_t)(m0 + nl) * lda;
  const _Float16* arow1 = arow0 + (size_t)16 * lda;
  const _Float16* brow = Btz + (size_t)(n0 + nl) * ldbt + hb * 8;
  const size_t bstep = (size_t)16 * ldbt;  // next N-subtile

  v16h Af[2][2];  // [stage][mi]
  v16h Bf[2][4];  // [stage][t]

  auto load_stage = [&](int p, int kb) {
    v8h a00 = *(const v8h*)(arow0 + kb + hb * 8);
    v8h a01 = *(const v8h*)(arow0 + kb + 16 + hb * 8);
    v8h a10 = *(const v8h*)(arow1 + kb + hb * 8);
    v8h a11 = *(const v8h*)(arow1 + kb + 16 + hb * 8);
#pragma unroll
    for (int q = 0; q < 8; ++q) {
      Af[p][0][q] = a00[q]; Af[p][0][q + 8] = a01[q];
      Af[p][1][q] = a10[q]; Af[p][1][q + 8] = a11[q];
    }
#pragma unroll
    for (int t = 0; t < 4; ++t) {
      const _Float16* bp = brow + (size_t)t * bstep + kb;
      v8h b0 = *(const v8h*)bp;
      v8h b1 = *(const v8h*)(bp + 16);
#pragma unroll
      for (int q = 0; q < 8; ++q) { Bf[p][t][q] = b0[q]; Bf[p][t][q + 8] = b1[q]; }
    }
  };
  auto compute_stage = [&](int p) {
#pragma unroll
    for (int t = 0; t < 4; ++t) {
      acc[0][t] = __builtin_amdgcn_wmma_f32_16x16x32_f16(
          false, Af[p][0], false, Bf[p][t], (short)0, acc[0][t], false, false);
      acc[1][t] = __builtin_amdgcn_wmma_f32_16x16x32_f16(
          false, Af[p][1], false, Bf[p][t], (short)0, acc[1][t], false, false);
    }
  };

  load_stage(0, 0);
  for (int kb = 0; kb < K; kb += 64) {
    load_stage(1, kb + 32);        // prefetch stage 1 before consuming stage 0
    compute_stage(0);
    if (kb + 64 < K) load_stage(0, kb + 64);  // prefetch next stage 0
    compute_stage(1);
  }

  float* Cfz = Cf ? Cf + z * cB : nullptr;
  _Float16* Chz = Ch ? Ch + z * chB : nullptr;
#pragma unroll
  for (int mi = 0; mi < 2; ++mi) {
    const int mbase = m0 + mi * 16 + 8 * hb;
#pragma unroll
    for (int t = 0; t < 4; ++t) {
      const int n = n0 + t * 16 + nl;
      const float bv = bias ? bias[n] : 0.0f;
      float vals[8];
#pragma unroll
      for (int r = 0; r < 8; ++r) {
        float v = acc[mi][t][r] + bv;
        if (act) v = gelu_exact(v);
        vals[r] = v;
      }
      if (Cfz) {
#pragma unroll
        for (int r = 0; r < 8; ++r)
          Cfz[(size_t)(mbase + r) * ldc + n] = vals[r];
      }
      if (Chz) {
        if (transCh) {  // store D transposed: contiguous 8 halves per lane
          v8h p;
#pragma unroll
          for (int r = 0; r < 8; ++r) p[r] = (_Float16)vals[r];
          *(v8h*)(Chz + (size_t)n * ldch + mbase) = p;
        } else {
#pragma unroll
          for (int r = 0; r < 8; ++r)
            Chz[(size_t)(mbase + r) * ldch + n] = (_Float16)vals[r];
        }
      }
    }
  }
}

// ---------------- softmax (one wave per row) ----------------
__global__ __launch_bounds__(32)
void k_softmax(const float* __restrict__ S, _Float16* __restrict__ P,
               int cols, float scale, int causal, int Lmod) {
  const int row = blockIdx.x;
  const float* s = S + (size_t)row * cols;
  _Float16* p = P + (size_t)row * cols;
  const int limit = causal ? ((row % Lmod) + 1) : cols;
  const int lane = threadIdx.x;
  float mx = -3.4e38f;
  for (int c = lane; c < limit; c += 32) mx = fmaxf(mx, s[c] * scale);
  mx = warp_max(mx);
  float sum = 0.0f;
  for (int c = lane; c < limit; c += 32) sum += __expf(s[c] * scale - mx);
  sum = warp_sum(sum);
  const float inv = 1.0f / sum;
  for (int c = lane; c < cols; c += 32) {
    float v = (c < limit) ? __expf(s[c] * scale - mx) * inv : 0.0f;
    p[c] = (_Float16)v;
  }
}

// ---------------- layernorm(+residual) (one wave per row) ----------------
__global__ __launch_bounds__(32)
void k_layernorm(const float* __restrict__ X, const float* __restrict__ R,
                 const float* __restrict__ g, const float* __restrict__ beta,
                 float* __restrict__ Y32, _Float16* __restrict__ Yh, int D) {
  const int row = blockIdx.x;
  const float* x = X + (size_t)row * D;
  const float* r = R ? R + (size_t)row * D : nullptr;
  const int lane = threadIdx.x;
  float s = 0.0f, ss = 0.0f;
  for (int c = lane; c < D; c += 32) {
    float v = x[c] + (r ? r[c] : 0.0f);
    s += v; ss += v * v;
  }
  s = warp_sum(s); ss = warp_sum(ss);
  const float mean = s / D;
  const float var = ss / D - mean * mean;
  const float rstd = rsqrtf(var + 1e-5f);
  for (int c = lane; c < D; c += 32) {
    float v = x[c] + (r ? r[c] : 0.0f);
    float y = (v - mean) * rstd * g[c] + beta[c];
    if (Y32) Y32[(size_t)row * D + c] = y;
    if (Yh) Yh[(size_t)row * D + c] = (_Float16)y;
  }
}

// ---------------- slice last PRED timesteps ----------------
__global__ void k_slice_out(const float* __restrict__ X, float* __restrict__ out) {
  int i = blockIdx.x * blockDim.x + threadIdx.x;
  const int total = kB * kPred * kCout;
  if (i >= total) return;
  int c = i % kCout;
  int t = (i / kCout) % kPred;
  int b = i / (kCout * kPred);
  out[i] = X[((size_t)(b * kLdec + (kLdec - kPred) + t)) * kCout + c];
}

// =======================================================================
extern "C" void kernel_launch(void* const* d_in, const int* in_sizes, int n_in,
                              void* d_out, int out_size, void* d_ws, size_t ws_size,
                              hipStream_t stream) {
  (void)in_sizes; (void)n_in; (void)out_size; (void)ws_size;
  int ii = 0;
  const float* x_dec     = (const float*)d_in[ii++];
  const float* x_enc     = (const float*)d_in[ii++];
  const float* enc_Wq    = (const float*)d_in[ii++];
  const float* enc_bq    = (const float*)d_in[ii++];
  const float* enc_Wk    = (const float*)d_in[ii++];
  const float* enc_bk    = (const float*)d_in[ii++];
  const float* enc_Wv    = (const float*)d_in[ii++];
  const float* enc_bv    = (const float*)d_in[ii++];
  const float* enc_Wo    = (const float*)d_in[ii++];
  const float* enc_bo    = (const float*)d_in[ii++];
  const float* enc_W1    = (const float*)d_in[ii++];
  const float* enc_b1    = (const float*)d_in[ii++];
  const float* enc_W2    = (const float*)d_in[ii++];
  const float* enc_b2    = (const float*)d_in[ii++];
  const float* enc_g1    = (const float*)d_in[ii++];
  const float* enc_beta1 = (const float*)d_in[ii++];
  const float* enc_g2    = (const float*)d_in[ii++];
  const float* enc_beta2 = (const float*)d_in[ii++];
  const float* enc_ng    = (const float*)d_in[ii++];
  const float* enc_nbeta = (const float*)d_in[ii++];
  const float* dec_sWq   = (const float*)d_in[ii++];
  const float* dec_sbq   = (const float*)d_in[ii++];
  const float* dec_sWk   = (const float*)d_in[ii++];
  const float* dec_sbk   = (const float*)d_in[ii++];
  const float* dec_sWv   = (const float*)d_in[ii++];
  const float* dec_sbv   = (const float*)d_in[ii++];
  const float* dec_sWo   = (const float*)d_in[ii++];
  const float* dec_sbo   = (const float*)d_in[ii++];
  const float* dec_cWq   = (const float*)d_in[ii++];
  const float* dec_cbq   = (const float*)d_in[ii++];
  const float* dec_cWk   = (const float*)d_in[ii++];
  const float* dec_cbk   = (const float*)d_in[ii++];
  const float* dec_cWv   = (const float*)d_in[ii++];
  const float* dec_cbv   = (const float*)d_in[ii++];
  const float* dec_cWo   = (const float*)d_in[ii++];
  const float* dec_cbo   = (const float*)d_in[ii++];
  const float* dec_W1    = (const float*)d_in[ii++];
  const float* dec_b1    = (const float*)d_in[ii++];
  const float* dec_W2    = (const float*)d_in[ii++];
  const float* dec_b2    = (const float*)d_in[ii++];
  const float* dec_g1    = (const float*)d_in[ii++];
  const float* dec_beta1 = (const float*)d_in[ii++];
  const float* dec_g2    = (const float*)d_in[ii++];
  const float* dec_beta2 = (const float*)d_in[ii++];
  const float* dec_g3    = (const float*)d_in[ii++];
  const float* dec_beta3 = (const float*)d_in[ii++];
  const float* dec_ng    = (const float*)d_in[ii++];
  const float* dec_nbeta = (const float*)d_in[ii++];
  const float* proj_W    = (const float*)d_in[ii++];
  const float* proj_b    = (const float*)d_in[ii++];

  // ---------------- workspace arena ----------------
  char* ws = (char*)d_ws;
  size_t off = 0;
  auto arena = [&](size_t bytes) -> char* {
    char* p = ws + off;
    off += (bytes + 255) & ~(size_t)255;
    return p;
  };
  const size_t rowsE = (size_t)kB * kLenc;  // 8192
  const size_t rowsD = (size_t)kB * kLdec;  // 4096
  float*    X32 = (float*)   arena(rowsE * kD * 4);          // residual stream f32
  _Float16* XH  = (_Float16*)arena(rowsE * kD * 2);          // activation f16
  _Float16* QH  = (_Float16*)arena(rowsE * kD * 2);
  _Float16* KH  = (_Float16*)arena(rowsE * kD * 2);
  _Float16* VT  = (_Float16*)arena((size_t)kD * rowsE * 2);  // V transposed [D, B*S]
  _Float16* AH  = (_Float16*)arena(rowsE * kD * 2);          // attn concat out
  float*    T32 = (float*)   arena(rowsE * kD * 4);          // gemm f32 out
  _Float16* FH  = (_Float16*)arena(rowsE * kDFF * 2);        // gelu hidden
  float*    SC  = (float*)   arena((size_t)kH * kLenc * kLenc * 4);   // scores (per batch)
  _Float16* PH  = (_Float16*)arena((size_t)kH * kLenc * kLenc * 2);   // probs
  _Float16* EH  = (_Float16*)arena(rowsE * kD * 2);          // encoder output f16
  _Float16* WT  = (_Float16*)arena((size_t)kDFF * kD * 2);   // transposed weight slot

  // ---------------- launch helpers ----------------
  auto cvt = [&](const float* s, _Float16* d, size_t n) {
    k_cvt_f16<<<dim3((unsigned)((n + 255) / 256)), 256, 0, stream>>>(s, d, (int)n);
  };
  auto cvtT = [&](const float* W, int K, int N) {
    size_t n = (size_t)K * N;
    k_cvt_transpose<<<dim3((unsigned)((n + 255) / 256)), 256, 0, stream>>>(W, WT, K, N);
  };
  auto gemm = [&](const _Float16* A, long long aB, int lda,
                  const _Float16* Bt, long long bB, int ldbt,
                  const float* bias,
                  float* Cf, long long cB, int ldc,
                  _Float16* Ch, long long chB, int ldch, int transCh,
                  int M, int N, int K, int act, int z) {
    dim3 grid(N / 64, M / 32, z);
    k_wmma_gemm<<<grid, 32, 0, stream>>>(A, aB, lda, Bt, bB, ldbt, bias,
                                         Cf, cB, ldc, Ch, chB, ldch, transCh, K, act);
  };
  auto layernorm = [&](const float* X, const float* R, const float* g, const float* b,
                       float* Y32, _Float16* Yh, int rows) {
    k_layernorm<<<dim3(rows), 32, 0, stream>>>(X, R, g, b, Y32, Yh, kD);
  };

  // Full multi-head attention: Q from Xq [B*Lq, D] f16, K/V from Xkv [B*Lkv, D] f16.
  // Writes concatenated head outputs to outAH [B*Lq, D] f16.
  auto attention = [&](const _Float16* Xq, int Lq, const _Float16* Xkv, int Lkv,
                       const float* Wq, const float* bq, const float* Wk, const float* bk,
                       const float* Wv, const float* bv, int causal, _Float16* outAH) {
    const int Mq = kB * Lq, Mkv = kB * Lkv;
    cvtT(Wq, kD, kD);
    gemm(Xq, 0, kD, WT, 0, kD, bq, nullptr, 0, 0, QH, 0, kD, 0, Mq, kD, kD, 0, 1);
    cvtT(Wk, kD, kD);
    gemm(Xkv, 0, kD, WT, 0, kD, bk, nullptr, 0, 0, KH, 0, kD, 0, Mkv, kD, kD, 0, 1);
    cvtT(Wv, kD, kD);
    // V written transposed: VT[d, b*Lkv + s], ld = Mkv
    gemm(Xkv, 0, kD, WT, 0, kD, bv, nullptr, 0, 0, VT, 0, Mkv, 1, Mkv, kD, kD, 0, 1);
    const float scale = 0.125f;  // 1/sqrt(64)
    for (int b = 0; b < kB; ++b) {
      // scores[h, l, s] = Q[b,l,h,:] . K[b,s,h,:]    (Bt = K itself, [N=s, K=dk])
      gemm(QH + (size_t)b * Lq * kD, 64, kD,
           KH + (size_t)b * Lkv * kD, 64, kD,
           nullptr,
           SC, (long long)Lq * Lkv, Lkv,
           nullptr, 0, 0, 0,
           Lq, Lkv, kDK, 0, kH);
      k_softmax<<<dim3(kH * Lq), 32, 0, stream>>>(SC, PH, Lkv, scale, causal, Lq);
      // out[h, l, d] = P[h,l,:] . V[b,:,h,d]   (Bt = VT head slice, ld = B*Lkv)
      gemm(PH, (long long)Lq * Lkv, Lkv,
           VT + (size_t)b * Lkv, (long long)64 * Mkv, Mkv,
           nullptr,
           nullptr, 0, 0,
           outAH + (size_t)b * Lq * kD, 64, kD, 0,
           Lq, kDK, Lkv, 0, kH);
    }
  };

  // ======================= ENCODER =======================
  cvt(x_enc, XH, rowsE * kD);
  const float* xcur = x_enc;  // f32 residual source
  for (int l = 0; l < kE; ++l) {
    const size_t wDD = (size_t)l * kD * kD;
    const size_t wDF = (size_t)l * kD * kDFF;
    attention(XH, kLenc, XH, kLenc,
              enc_Wq + wDD, enc_bq + l * kD, enc_Wk + wDD, enc_bk + l * kD,
              enc_Wv + wDD, enc_bv + l * kD, 0, AH);
    cvtT(enc_Wo + wDD, kD, kD);
    gemm(AH, 0, kD, WT, 0, kD, enc_bo + l * kD, T32, 0, kD,
         nullptr, 0, 0, 0, (int)rowsE, kD, kD, 0, 1);
    layernorm(xcur, T32, enc_g1 + l * kD, enc_beta1 + l * kD, X32, XH, (int)rowsE);
    xcur = X32;
    // FFN
    cvtT(enc_W1 + wDF, kD, kDFF);
    gemm(XH, 0, kD, WT, 0, kD, enc_b1 + l * kDFF, nullptr, 0, 0,
         FH, 0, kDFF, 0, (int)rowsE, kDFF, kD, 1 /*gelu*/, 1);
    cvtT(enc_W2 + wDF, kDFF, kD);
    gemm(FH, 0, kDFF, WT, 0, kDFF, enc_b2 + l * kD, T32, 0, kD,
         nullptr, 0, 0, 0, (int)rowsE, kD, kDFF, 0, 1);
    layernorm(X32, T32, enc_g2 + l * kD, enc_beta2 + l * kD, X32, XH, (int)rowsE);
  }
  layernorm(X32, nullptr, enc_ng, enc_nbeta, nullptr, EH, (int)rowsE);  // enc_out -> EH

  // ======================= DECODER =======================
  cvt(x_dec, XH, rowsD * kD);
  // causal self-attention
  attention(XH, kLdec, XH, kLdec,
            dec_sWq, dec_sbq, dec_sWk, dec_sbk, dec_sWv, dec_sbv, 1, AH);
  cvtT(dec_sWo, kD, kD);
  gemm(AH, 0, kD, WT, 0, kD, dec_sbo, T32, 0, kD,
       nullptr, 0, 0, 0, (int)rowsD, kD, kD, 0, 1);
  layernorm(x_dec, T32, dec_g1, dec_beta1, X32, XH, (int)rowsD);
  // cross-attention (K/V from encoder output)
  attention(XH, kLdec, EH, kLenc,
            dec_cWq, dec_cbq, dec_cWk, dec_cbk, dec_cWv, dec_cbv, 0, AH);
  cvtT(dec_cWo, kD, kD);
  gemm(AH, 0, kD, WT, 0, kD, dec_cbo, T32, 0, kD,
       nullptr, 0, 0, 0, (int)rowsD, kD, kD, 0, 1);
  layernorm(X32, T32, dec_g2, dec_beta2, X32, XH, (int)rowsD);
  // FFN
  cvtT(dec_W1, kD, kDFF);
  gemm(XH, 0, kD, WT, 0, kD, dec_b1, nullptr, 0, 0,
       FH, 0, kDFF, 0, (int)rowsD, kDFF, kD, 1 /*gelu*/, 1);
  cvtT(dec_W2, kDFF, kD);
  gemm(FH, 0, kDFF, WT, 0, kDFF, dec_b2, T32, 0, kD,
       nullptr, 0, 0, 0, (int)rowsD, kD, kDFF, 0, 1);
  layernorm(X32, T32, dec_g3, dec_beta3, X32, nullptr, (int)rowsD);
  layernorm(X32, nullptr, dec_ng, dec_nbeta, nullptr, XH, (int)rowsD);
  // projection D -> C_OUT
  cvtT(proj_W, kD, kCout);
  gemm(XH, 0, kD, WT, 0, kD, proj_b, T32, 0, kCout,
       nullptr, 0, 0, 0, (int)rowsD, kCout, kD, 0, 1);
  // slice last PRED timesteps
  k_slice_out<<<dim3((kB * kPred * kCout + 255) / 256), 256, 0, stream>>>(T32, (float*)d_out);
}
